// DeformableConv_6339371729442
// MI455X (gfx1250) — matrix operand order repaired and checked by reading, not compiled, not run on hardware
//
#include <hip/hip_runtime.h>
#include <math.h>

#define B_   2
#define T1_  243
#define T2_  9
#define T3_  27
#define P_   17
#define K2_  9
#define PK_  153              // P*K2
#define C_   128
#define NPTQ (B_*T1_*PK_)     // 74358 sample points (b,t1,q)
#define NPTV (B_*T1_*P_)      // 8262  value points  (b,t1,p)
#define ALPHA_ 0.9f

#define EMB_CHUNKS  23        // ceil(48/32)+ceil(96/32)+ceil(192/32)+ceil(384/32)
#define CONV_CHUNKS 36        // 1152/32
#define FRAG_ELEMS  512       // 32 lanes x 16 bf16 per (chunk, ntile) block

typedef __attribute__((ext_vector_type(16))) __bf16 bf16x16;
typedef __attribute__((ext_vector_type(8)))  float  fx8;

__device__ __forceinline__ fx8 wmma_bf16(bf16x16 a, bf16x16 b, fx8 c) {
  // D = A(16x32 bf16) * B(32x16 bf16) + C(16x16 f32)
  return __builtin_amdgcn_wmma_f32_16x16x32_bf16(
      false, a, false, b, (short)0, c, false, false);
}

__device__ __forceinline__ float wave_sum(float v) {
#pragma unroll
  for (int o = 16; o > 0; o >>= 1) v += __shfl_xor(v, o, 32);
  return v;
}

// ---------------------------------------------------------------------------
// aw = softmax(value @ w_attn^T + b_attn) over the 2-vector, one wave / point
// ---------------------------------------------------------------------------
__global__ void k_attn(const float* __restrict__ value,
                       const float* __restrict__ w_attn,
                       const float* __restrict__ b_attn,
                       float* __restrict__ aw) {
  int wave = threadIdx.x >> 5, lane = threadIdx.x & 31;
  int pt = blockIdx.x * 8 + wave;
  if (pt >= NPTV) return;
  const float* v = value + (size_t)pt * C_;
  float s0 = 0.f, s1 = 0.f;
  for (int c = lane; c < C_; c += 32) {
    float x = v[c];
    s0 += x * w_attn[c];
    s1 += x * w_attn[C_ + c];
  }
  s0 = wave_sum(s0);
  s1 = wave_sum(s1);
  if (lane == 0) {
    s0 += b_attn[0]; s1 += b_attn[1];
    float m = fmaxf(s0, s1);
    float e0 = expf(s0 - m), e1 = expf(s1 - m);
    float inv = 1.f / (e0 + e1);
    aw[pt * 2 + 0] = e0 * inv;
    aw[pt * 2 + 1] = e1 * inv;
  }
}

// ---------------------------------------------------------------------------
// sampling_positions = key + tanh(query @ w_off^T + b_off) * tiled(aw)
// ---------------------------------------------------------------------------
__global__ void k_samp(const float* __restrict__ query,
                       const float* __restrict__ key,
                       const float* __restrict__ w_off,
                       const float* __restrict__ b_off,
                       const float* __restrict__ aw,
                       float* __restrict__ sampos) {
  int wave = threadIdx.x >> 5, lane = threadIdx.x & 31;
  int pt = blockIdx.x * 8 + wave;
  if (pt >= NPTQ) return;
  const float* q = query + (size_t)pt * C_;
  float s0 = 0.f, s1 = 0.f;
  for (int c = lane; c < C_; c += 32) {
    float x = q[c];
    s0 += x * w_off[c];
    s1 += x * w_off[C_ + c];
  }
  s0 = wave_sum(s0);
  s1 = wave_sum(s1);
  if (lane == 0) {
    float o0 = tanhf(s0 + b_off[0]);
    float o1 = tanhf(s1 + b_off[1]);
    int qq = pt % PK_;
    int awi = (pt / PK_) * P_ + (qq % P_);   // jnp.tile: q -> p = q % P
    sampos[pt * 2 + 0] = key[pt * 2 + 0] + o0 * aw[awi * 2 + 0];
    sampos[pt * 2 + 1] = key[pt * 2 + 1] + o1 * aw[awi * 2 + 1];
  }
}

// ---------------------------------------------------------------------------
// Pre-pack all B matrices as bf16 in WMMA per-lane fragment order:
//   block (chunk, ntile) = 32 lanes x 16 bf16; lane l owns col n = ntile*16
//   + (l&15), element j -> K = (l>>4)*16 + j   (contiguous k order, same as A).
// gBemb : 23 chunks (levels zero-padded to 32-K) x 8 ntiles, B[k][n]=we[n][k]
// gBconv: 36 chunks x 8 ntiles, B[k*C+c][o] = conv_w[o][c][k]
// ---------------------------------------------------------------------------
__global__ void k_prep(const float* __restrict__ we0,
                       const float* __restrict__ we1,
                       const float* __restrict__ we2,
                       const float* __restrict__ we3,
                       const float* __restrict__ conv_w,
                       __bf16* __restrict__ gBemb,
                       __bf16* __restrict__ gBconv) {
  const float* wemb[4] = {we0, we1, we2, we3};
  const int cArr[4] = {48, 96, 192, 384};
  const int totE = EMB_CHUNKS * 8 * FRAG_ELEMS;
  const int tot = (EMB_CHUNKS + CONV_CHUNKS) * 8 * FRAG_ELEMS;
  for (int t = blockIdx.x * blockDim.x + threadIdx.x; t < tot;
       t += gridDim.x * blockDim.x) {
    if (t < totE) {
      int j = t & 15, lane = (t >> 4) & 31, ntile = (t >> 9) & 7;
      int chunk = t >> 12;
      int lv, kc;
      if (chunk < 2)       { lv = 0; kc = chunk; }
      else if (chunk < 5)  { lv = 1; kc = chunk - 2; }
      else if (chunk < 11) { lv = 2; kc = chunk - 5; }
      else                 { lv = 3; kc = chunk - 11; }
      int c = cArr[lv];
      int n = ntile * 16 + (lane & 15);
      int kk = ((lane >> 4) << 4) + j;
      int ch = kc * 32 + kk;
      float v = (ch < c) ? wemb[lv][(size_t)n * c + ch] : 0.f;
      gBemb[t] = (__bf16)v;
    } else {
      int u = t - totE;
      int j = u & 15, lane = (u >> 4) & 31, ntile = (u >> 9) & 7;
      int chunk = u >> 12;
      int o = ntile * 16 + (lane & 15);
      int kk = ((lane >> 4) << 4) + j;
      int kidx = chunk * 32 + kk;            // kidx = k*C + c
      int k = kidx >> 7, cc = kidx & 127;
      gBconv[u] = (__bf16)conv_w[((size_t)o * C_ + cc) * K2_ + k];
    }
  }
}

// ---------------------------------------------------------------------------
// new_query: bilinear gather -> LDS (bf16, A-fragment swizzled order) ->
// WMMA against prepacked global B fragments; 4 levels accumulated, mean,
// blend with query.  16-point M tile / block, 4 waves x 2 N tiles.
// ---------------------------------------------------------------------------
__global__ void __launch_bounds__(128) k_main(
    const float* __restrict__ f0, const float* __restrict__ f1,
    const float* __restrict__ f2, const float* __restrict__ f3,
    const __bf16* __restrict__ gBemb,
    const float* __restrict__ be0, const float* __restrict__ be1,
    const float* __restrict__ be2, const float* __restrict__ be3,
    const float* __restrict__ query, const float* __restrict__ sampos,
    float* __restrict__ new_query) {
  __shared__ float  s_gx[16], s_gy[16];
  __shared__ int    s_n[16], s_ok[16];
  __shared__ __attribute__((aligned(32))) __bf16 sAf[FRAG_ELEMS];

  const float* fmap[4] = {f0, f1, f2, f3};
  const int cArr[4] = {48, 96, 192, 384};
  const int hArr[4] = {96, 48, 24, 12};
  const int wArr[4] = {72, 36, 18, 9};

  int tid = threadIdx.x;
  int lane = tid & 31, wave = tid >> 5;
  int base = blockIdx.x * 16;

  if (tid < 16) {
    int gp = base + tid;
    if (gp < NPTQ) {
      s_ok[tid] = 1;
      s_gx[tid] = sampos[gp * 2 + 0];
      s_gy[tid] = sampos[gp * 2 + 1];
      int bt = gp / PK_;                 // b*T1 + t1
      int b = bt / T1_, t1 = bt % T1_;
      s_n[tid] = b * T2_ + t1 / T3_;     // feature-map batch index
    } else {
      s_ok[tid] = 0; s_gx[tid] = 0.f; s_gy[tid] = 0.f; s_n[tid] = 0;
    }
  }

  fx8 acc0 = {};
  fx8 acc1 = {};
  int chunkG = 0;

  for (int lv = 0; lv < 4; ++lv) {
    const int c = cArr[lv], H = hArr[lv], W = wArr[lv];
    const float* f = fmap[lv];
    const int nch = (c + 31) >> 5;       // K chunks of 32 (zero padded)
    for (int kc = 0; kc < nch; ++kc) {
      __syncthreads();
      // stage A: bilinear-gathered 16x32 tile, written in fragment order:
      //   lane = m + (k>>4)*16 ; j = k&15
      for (int e = tid; e < 512; e += 128) {
        int m = e >> 5, kk = e & 31;
        int ch = kc * 32 + kk;
        float val = 0.f;
        if (ch < c && s_ok[m]) {
          float gx = s_gx[m], gy = s_gy[m];
          float ix = ((gx + 1.f) * W - 1.f) * 0.5f;
          float iy = ((gy + 1.f) * H - 1.f) * 0.5f;
          float fx0 = floorf(ix), fy0 = floorf(iy);
          int x0 = (int)fx0, y0 = (int)fy0;
          float wx1 = ix - fx0, wy1 = iy - fy0;
          float wx0 = 1.f - wx1, wy0 = 1.f - wy1;
          const float* fc = f + (size_t)(s_n[m] * c + ch) * H * W;
          float v00 = 0.f, v01 = 0.f, v10 = 0.f, v11 = 0.f;
          if (y0 >= 0 && y0 < H) {
            if (x0 >= 0 && x0 < W)         v00 = fc[y0 * W + x0];
            if (x0 + 1 >= 0 && x0 + 1 < W) v01 = fc[y0 * W + x0 + 1];
          }
          if (y0 + 1 >= 0 && y0 + 1 < H) {
            if (x0 >= 0 && x0 < W)         v10 = fc[(y0 + 1) * W + x0];
            if (x0 + 1 >= 0 && x0 + 1 < W) v11 = fc[(y0 + 1) * W + x0 + 1];
          }
          val = v00 * wy0 * wx0 + v01 * wy0 * wx1 +
                v10 * wy1 * wx0 + v11 * wy1 * wx1;
        }
        int flane = m + ((kk >> 4) << 4);
        int fj = kk & 15;
        sAf[flane * 16 + fj] = (__bf16)val;
      }
      __syncthreads();
      // one 32B LDS vector load for A; two 32B global (L2-hit) loads for B
      bf16x16 afrag = *(const bf16x16*)&sAf[lane * 16];
      const __bf16* bbase =
          gBemb + ((size_t)(chunkG * 8 + wave * 2) * FRAG_ELEMS) + lane * 16;
      bf16x16 bfrag0 = *(const bf16x16*)bbase;
      bf16x16 bfrag1 = *(const bf16x16*)(bbase + FRAG_ELEMS);
      if (chunkG + 1 < EMB_CHUNKS)
        __builtin_prefetch(bbase + 8 * FRAG_ELEMS, 0, 3);
      acc0 = wmma_bf16(afrag, bfrag0, acc0);
      acc1 = wmma_bf16(afrag, bfrag1, acc1);
      ++chunkG;
    }
  }

  // epilogue: mean over 4 levels (+ bias sum), blend with query
  int ncolA = wave * 32 + (lane & 15);
  int ncolB = ncolA + 16;
  float biasA = 0.25f * (be0[ncolA] + be1[ncolA] + be2[ncolA] + be3[ncolA]);
  float biasB = 0.25f * (be0[ncolB] + be1[ncolB] + be2[ncolB] + be3[ncolB]);
#pragma unroll
  for (int r = 0; r < 8; ++r) {
    int m = ((lane >> 4) << 3) + r;
    int gp = base + m;
    if (gp < NPTQ) {
      float qa = query[(size_t)gp * C_ + ncolA];
      float qb = query[(size_t)gp * C_ + ncolB];
      new_query[(size_t)gp * C_ + ncolA] =
          (1.f - ALPHA_) * (0.25f * acc0[r] + biasA) + ALPHA_ * qa;
      new_query[(size_t)gp * C_ + ncolB] =
          (1.f - ALPHA_) * (0.25f * acc1[r] + biasB) + ALPHA_ * qb;
    }
  }
}

// ---------------------------------------------------------------------------
// new_value[b,t,p,o] = sum_{k,c} new_query[b,t,p*K2+k,c] * conv_w[o,c,k] + b
// GEMM [8262 x 1152] x [1152 x 128]; B prepacked, A staged swizzled in LDS.
// ---------------------------------------------------------------------------
__global__ void __launch_bounds__(128) k_conv(
    const float* __restrict__ new_query, const __bf16* __restrict__ gBconv,
    const float* __restrict__ conv_b, float* __restrict__ new_value) {
  __shared__ __attribute__((aligned(32))) __bf16 sAf[FRAG_ELEMS];
  int tid = threadIdx.x, lane = tid & 31, wave = tid >> 5;
  int base = blockIdx.x * 16;
  fx8 acc0 = {};
  fx8 acc1 = {};

  for (int kc = 0; kc < CONV_CHUNKS; ++kc) {
    __syncthreads();
    for (int e = tid; e < 512; e += 128) {
      int m = e >> 5, kk = e & 31;
      int gp = base + m;
      float val = 0.f;
      if (gp < NPTV) {
        int p = gp % P_;
        int bt = gp / P_;
        size_t rowbase = ((size_t)bt * PK_ + (size_t)p * K2_) * C_;
        val = new_query[rowbase + kc * 32 + kk];   // (k,c) contiguous
      }
      int flane = m + ((kk >> 4) << 4);
      int fj = kk & 15;
      sAf[flane * 16 + fj] = (__bf16)val;
    }
    __syncthreads();
    bf16x16 afrag = *(const bf16x16*)&sAf[lane * 16];
    const __bf16* bbase =
        gBconv + ((size_t)(kc * 8 + wave * 2) * FRAG_ELEMS) + lane * 16;
    bf16x16 bfrag0 = *(const bf16x16*)bbase;
    bf16x16 bfrag1 = *(const bf16x16*)(bbase + FRAG_ELEMS);
    if (kc + 1 < CONV_CHUNKS)
      __builtin_prefetch(bbase + 8 * FRAG_ELEMS, 0, 3);
    acc0 = wmma_bf16(afrag, bfrag0, acc0);
    acc1 = wmma_bf16(afrag, bfrag1, acc1);
  }

  int ncolA = wave * 32 + (lane & 15);
  int ncolB = ncolA + 16;
#pragma unroll
  for (int r = 0; r < 8; ++r) {
    int gp = base + ((lane >> 4) << 3) + r;
    if (gp < NPTV) {
      new_value[(size_t)gp * C_ + ncolA] = acc0[r] + conv_b[ncolA];
      new_value[(size_t)gp * C_ + ncolB] = acc1[r] + conv_b[ncolB];
    }
  }
}

// ---------------------------------------------------------------------------
extern "C" void kernel_launch(void* const* d_in, const int* in_sizes, int n_in,
                              void* d_out, int out_size, void* d_ws,
                              size_t ws_size, hipStream_t stream) {
  (void)in_sizes; (void)n_in; (void)out_size; (void)ws_size;
  const float* f0     = (const float*)d_in[0];
  const float* f1     = (const float*)d_in[1];
  const float* f2     = (const float*)d_in[2];
  const float* f3     = (const float*)d_in[3];
  const float* query  = (const float*)d_in[4];
  const float* key    = (const float*)d_in[5];
  const float* value  = (const float*)d_in[6];
  const float* w_off  = (const float*)d_in[7];
  const float* b_off  = (const float*)d_in[8];
  const float* w_attn = (const float*)d_in[9];
  const float* b_attn = (const float*)d_in[10];
  const float* we0    = (const float*)d_in[11];
  const float* be0    = (const float*)d_in[12];
  const float* we1    = (const float*)d_in[13];
  const float* be1    = (const float*)d_in[14];
  const float* we2    = (const float*)d_in[15];
  const float* be2    = (const float*)d_in[16];
  const float* we3    = (const float*)d_in[17];
  const float* be3    = (const float*)d_in[18];
  const float* conv_w = (const float*)d_in[19];
  const float* conv_b = (const float*)d_in[20];

  float* out       = (float*)d_out;
  float* new_query = out;                                     // B*T1*PK*C
  float* new_value = new_query + (size_t)B_ * T1_ * PK_ * C_; // B*T1*P*C
  float* sampos    = new_value + (size_t)B_ * T1_ * P_ * C_;  // N*T3*PK*2

  // workspace layout (bytes): gBemb | gBconv | aw
  char* ws = (char*)d_ws;
  __bf16* gBemb  = (__bf16*)ws;                               // 23*8*512 bf16
  __bf16* gBconv = gBemb + (size_t)EMB_CHUNKS * 8 * FRAG_ELEMS;
  float*  aw     = (float*)(gBconv + (size_t)CONV_CHUNKS * 8 * FRAG_ELEMS);

  k_attn<<<(NPTV + 7) / 8, 256, 0, stream>>>(value, w_attn, b_attn, aw);
  k_samp<<<(NPTQ + 7) / 8, 256, 0, stream>>>(query, key, w_off, b_off, aw,
                                             sampos);
  k_prep<<<((EMB_CHUNKS + CONV_CHUNKS) * 8 * FRAG_ELEMS + 255) / 256, 256, 0,
           stream>>>(we0, we1, we2, we3, conv_w, gBemb, gBconv);
  k_main<<<(NPTQ + 15) / 16, 128, 0, stream>>>(
      f0, f1, f2, f3, gBemb, be0, be1, be2, be3, query, sampos, new_query);
  k_conv<<<(NPTV + 15) / 16, 128, 0, stream>>>(new_query, gBconv, conv_b,
                                               new_value);
}
